// BAMM_67130338836934
// MI455X (gfx1250) — matrix operand order repaired and checked by reading, not compiled, not run on hardware
//
#include <hip/hip_runtime.h>
#include <hip/hip_bf16.h>

// ---------------------------------------------------------------------------
// Problem constants (from reference): B=8, C=512, Ck=128, H=W=128, h=w=32,
// N = h*w = 1024, DS=4.
// ---------------------------------------------------------------------------
#define BB   8
#define CC   512
#define CKK  128
#define HH   128
#define WW   128
#define NN   1024   // 32*32 pooled pixels

typedef __attribute__((ext_vector_type(16))) __bf16 v16bf;
typedef __attribute__((ext_vector_type(8)))  __bf16 v8bf;
typedef __attribute__((ext_vector_type(8)))  float  v8f;

// ---------------------------------------------------------------------------
// Fragment loaders following the CDNA5 WMMA VGPR layouts (ISA 7.12.2).
//
// A (16x32, MxK), 16-bit: lane L -> row m = L%16, hi = L/16.
//   VGPR j<4 half h : K = 8*hi + 2j + h   -> elements e=0..7  : K = 8*hi + e
//   VGPR j>=4       : K = 16 + 8*hi + ... -> elements e=8..15 : K = 16+8*hi+(e-8)
// Two contiguous 16B loads per lane.
// ---------------------------------------------------------------------------
__device__ __forceinline__ v16bf load_a_frag(const __bf16* __restrict__ Arow,
                                             int k0, int hi) {
    const v8bf p0 = *(const v8bf*)(Arow + k0 + 8 * hi);
    const v8bf p1 = *(const v8bf*)(Arow + k0 + 16 + 8 * hi);
    v16bf a;
#pragma unroll
    for (int e = 0; e < 8; ++e) { a[e] = p0[e]; a[e + 8] = p1[e]; }
    return a;
}

// ---------------------------------------------------------------------------
// B (32x16, KxN), 16-bit: lane L -> col n = L%16, hi = L/16.
//   element e<8 : K = 8*hi + e ; e>=8 : K = 16 + 8*hi + (e-8)
// Source stored K-major per column ("Bt", row-major [Ncols, K]) so this is
// two contiguous 16B loads per lane.
// ---------------------------------------------------------------------------
__device__ __forceinline__ v16bf load_b_frag(const __bf16* __restrict__ Btrow,
                                             int k0, int hi) {
    const v8bf p0 = *(const v8bf*)(Btrow + k0 + 8 * hi);
    const v8bf p1 = *(const v8bf*)(Btrow + k0 + 16 + 8 * hi);
    v16bf b;
#pragma unroll
    for (int e = 0; e < 8; ++e) { b[e] = p0[e]; b[e + 8] = p1[e]; }
    return b;
}

// ---------------------------------------------------------------------------
// Generic batched GEMM:  C[b][m][n] = sum_k A[b][m][k] * Bt[b][n][k] + bias[m]
//   A  : bf16, row-major [M, lda]   (strideA elements per batch; 0 = shared)
//   Bt : bf16, row-major [N, ldb]   (i.e. B stored transposed / K-major)
//   storeMode: 0 = f32 row-major [M, ldc]
//              1 = bf16 row-major [M, ldc]
//              2 = bf16 TRANSPOSED store: Cout[n*ldc + m]
// Block = 256 threads = 8 waves; wave w -> rows [blockIdx.y*128 + 16w, +16),
// cols [blockIdx.x*64, +64) as 4 accumulators sharing one A fragment.
// ---------------------------------------------------------------------------
__global__ __launch_bounds__(256)
void wmma_gemm_bf16(const __bf16* __restrict__ A,  long long strideA, int lda,
                    const __bf16* __restrict__ Bt, long long strideB, int ldb,
                    const float* __restrict__ bias,
                    void* __restrict__ Cout, long long strideC, int ldc,
                    int storeMode, int K)
{
    const int lane = threadIdx.x & 31;
    const int wave = threadIdx.x >> 5;
    const int hi   = lane >> 4;
    const int lo   = lane & 15;
    const int bz   = blockIdx.z;
    const int m0   = blockIdx.y * 128 + wave * 16;
    const int n0   = blockIdx.x * 64;

    const __bf16* Arow = A  + (size_t)bz * strideA + (size_t)(m0 + lo) * lda;
    const __bf16* Bb   = Bt + (size_t)bz * strideB;
    const __bf16* Brow0 = Bb + (size_t)(n0 +  0 + lo) * ldb;
    const __bf16* Brow1 = Bb + (size_t)(n0 + 16 + lo) * ldb;
    const __bf16* Brow2 = Bb + (size_t)(n0 + 32 + lo) * ldb;
    const __bf16* Brow3 = Bb + (size_t)(n0 + 48 + lo) * ldb;

    v8f acc[4] = {};

    for (int k0 = 0; k0 < K; k0 += 32) {
        // speculative prefetch of the next K-tile (global_prefetch_b8)
        __builtin_prefetch(Arow + k0 + 32, 0, 1);
        __builtin_prefetch(Brow0 + k0 + 32, 0, 1);

        v16bf a  = load_a_frag(Arow, k0, hi);
        v16bf b0 = load_b_frag(Brow0, k0, hi);
        v16bf b1 = load_b_frag(Brow1, k0, hi);
        v16bf b2 = load_b_frag(Brow2, k0, hi);
        v16bf b3 = load_b_frag(Brow3, k0, hi);

        acc[0] = __builtin_amdgcn_wmma_f32_16x16x32_bf16(
            false, a, false, b0, (short)0, acc[0], false, false);
        acc[1] = __builtin_amdgcn_wmma_f32_16x16x32_bf16(
            false, a, false, b1, (short)0, acc[1], false, false);
        acc[2] = __builtin_amdgcn_wmma_f32_16x16x32_bf16(
            false, a, false, b2, (short)0, acc[2], false, false);
        acc[3] = __builtin_amdgcn_wmma_f32_16x16x32_bf16(
            false, a, false, b3, (short)0, acc[3], false, false);
    }

    // D layout (f32 16x16): VGPR r, lane L -> m = 8*(L/16) + r, n = L%16.
    float brs[8];
#pragma unroll
    for (int r = 0; r < 8; ++r)
        brs[r] = bias ? bias[m0 + 8 * hi + r] : 0.0f;

#pragma unroll
    for (int t = 0; t < 4; ++t) {
#pragma unroll
        for (int r = 0; r < 8; ++r) {
            const int m = m0 + 8 * hi + r;
            const int n = n0 + t * 16 + lo;
            const float val = acc[t][r] + brs[r];
            if (storeMode == 0) {
                ((float*)Cout)[(size_t)bz * strideC + (size_t)m * ldc + n] = val;
            } else if (storeMode == 1) {
                ((__bf16*)Cout)[(size_t)bz * strideC + (size_t)m * ldc + n] =
                    (__bf16)val;
            } else {
                ((__bf16*)Cout)[(size_t)bz * strideC + (size_t)n * ldc + m] =
                    (__bf16)val;
            }
        }
    }
}

// ---------------------------------------------------------------------------
// 4x4 average pool of [B,C,128,128] f32 -> TRANSPOSED bf16 [B, N=1024, C=512]
// (K-major layout for the GEMM B-operand). One thread per pooled element.
// ---------------------------------------------------------------------------
__global__ __launch_bounds__(256)
void pool_to_bf16T(const float* __restrict__ src, __bf16* __restrict__ dstT)
{
    const size_t tid = (size_t)blockIdx.x * 256 + threadIdx.x; // B*C*N threads
    const int n = (int)(tid & (NN - 1));
    const int c = (int)((tid >> 10) & (CC - 1));
    const int b = (int)(tid >> 19);
    const int i = n >> 5, j = n & 31;

    const float4* p = (const float4*)
        (src + ((((size_t)b * CC + c) * HH + i * 4) * WW + j * 4));
    float s = 0.0f;
#pragma unroll
    for (int r = 0; r < 4; ++r) {           // row stride = 128 floats = 32 float4
        const float4 v = p[r * 32];
        s += v.x + v.y + v.z + v.w;
    }
    dstT[((size_t)b * NN + n) * CC + c] = (__bf16)(s * 0.0625f);
}

// ---------------------------------------------------------------------------
// f32 -> bf16 (weights)
// ---------------------------------------------------------------------------
__global__ __launch_bounds__(256)
void f32_to_bf16(const float* __restrict__ s, __bf16* __restrict__ d, int n)
{
    const int i = blockIdx.x * 256 + threadIdx.x;
    if (i < n) d[i] = (__bf16)s[i];
}

// ---------------------------------------------------------------------------
// Row softmax of energy (f32 [rows,1024]) with scale folded in; bf16 output
// [rows,1024] (this is exactly the Bt layout the out-GEMM wants).
// One 256-thread block per row; 4 elements per thread.
// ---------------------------------------------------------------------------
__global__ __launch_bounds__(256)
void softmax_row(const float* __restrict__ energy, __bf16* __restrict__ attn,
                 float scale)
{
    __shared__ float red[256];
    const int row = blockIdx.x;
    const int t   = threadIdx.x;

    float4 v = ((const float4*)(energy + (size_t)row * NN))[t];
    v.x *= scale; v.y *= scale; v.z *= scale; v.w *= scale;

    red[t] = fmaxf(fmaxf(v.x, v.y), fmaxf(v.z, v.w));
    __syncthreads();
    for (int s = 128; s > 0; s >>= 1) {
        if (t < s) red[t] = fmaxf(red[t], red[t + s]);
        __syncthreads();
    }
    const float rmax = red[0];
    __syncthreads();

    const float e0 = __expf(v.x - rmax), e1 = __expf(v.y - rmax);
    const float e2 = __expf(v.z - rmax), e3 = __expf(v.w - rmax);
    red[t] = e0 + e1 + e2 + e3;
    __syncthreads();
    for (int s = 128; s > 0; s >>= 1) {
        if (t < s) red[t] += red[t + s];
        __syncthreads();
    }
    const float inv = 1.0f / red[0];

    __bf16* dst = attn + (size_t)row * NN + t * 4;
    dst[0] = (__bf16)(e0 * inv);
    dst[1] = (__bf16)(e1 * inv);
    dst[2] = (__bf16)(e2 * inv);
    dst[3] = (__bf16)(e3 * inv);
}

// ---------------------------------------------------------------------------
// Nearest-neighbor x4 upsample of pooled out [B,C,32,32] (f32 [B,C,N]) + c2.
// One thread per float4 along W; all 4 lanes share one pooled value.
// ---------------------------------------------------------------------------
__global__ __launch_bounds__(256)
void upsample_add(const float* __restrict__ outp, const float* __restrict__ c2,
                  float* __restrict__ out)
{
    const size_t tid  = (size_t)blockIdx.x * 256 + threadIdx.x; // B*C*H*W/4
    const size_t base = tid * 4;
    const int x4 = (int)(base & (WW - 1));
    size_t rem   = base >> 7;
    const int y  = (int)(rem & (HH - 1)); rem >>= 7;
    const int c  = (int)(rem & (CC - 1));
    const int b  = (int)(rem >> 9);

    const int n = (y >> 2) * 32 + (x4 >> 2);
    const float pv = outp[(((size_t)b * CC + c) << 10) + n];
    const float4 cv = *(const float4*)(c2 + base);
    float4 o;
    o.x = pv + cv.x; o.y = pv + cv.y; o.z = pv + cv.z; o.w = pv + cv.w;
    *(float4*)(out + base) = o;
}

// ---------------------------------------------------------------------------
// Host-side launch
// ---------------------------------------------------------------------------
extern "C" void kernel_launch(void* const* d_in, const int* in_sizes, int n_in,
                              void* d_out, int out_size, void* d_ws, size_t ws_size,
                              hipStream_t stream)
{
    (void)in_sizes; (void)n_in; (void)out_size; (void)ws_size;

    const float* input = (const float*)d_in[0];
    const float* c2    = (const float*)d_in[1];
    const float* Wq    = (const float*)d_in[2];
    const float* bq    = (const float*)d_in[3];
    const float* Wk    = (const float*)d_in[4];
    const float* bk    = (const float*)d_in[5];
    const float* Wv    = (const float*)d_in[6];
    const float* bv    = (const float*)d_in[7];
    float* out = (float*)d_out;

    // ---- workspace carve-up (256B aligned), ~97 MB total ----
    char* ws = (char*)d_ws;
    size_t off = 0;
    auto carve = [&](size_t bytes) -> char* {
        char* p = ws + off;
        off += (bytes + 255) & ~(size_t)255;
        return p;
    };
    __bf16* xbT  = (__bf16*)carve((size_t)BB * NN * CC  * 2); // pooled input^T
    __bf16* ybT  = (__bf16*)carve((size_t)BB * NN * CC  * 2); // pooled c2^T
    __bf16* Wqb  = (__bf16*)carve((size_t)CKK * CC      * 2);
    __bf16* Wkb  = (__bf16*)carve((size_t)CKK * CC      * 2);
    __bf16* Wvb  = (__bf16*)carve((size_t)CC  * CC      * 2);
    __bf16* qT   = (__bf16*)carve((size_t)BB * NN * CKK * 2); // q^T [B,N,Ck]
    __bf16* kT   = (__bf16*)carve((size_t)BB * NN * CKK * 2); // k^T [B,N,Ck]
    __bf16* vB   = (__bf16*)carve((size_t)BB * CC * NN  * 2); // v   [B,C,N]
    float*  ener = (float*) carve((size_t)BB * NN * NN  * 4); // energy f32
    __bf16* attn = (__bf16*)carve((size_t)BB * NN * NN  * 2); // softmax bf16
    float*  outp = (float*) carve((size_t)BB * CC * NN  * 4); // out pooled f32

    // 1) fused avg-pool + bf16 convert (transposed for GEMM B-operand)
    pool_to_bf16T<<<(BB * CC * NN) / 256, 256, 0, stream>>>(input, xbT);
    pool_to_bf16T<<<(BB * CC * NN) / 256, 256, 0, stream>>>(c2, ybT);

    // 2) weight conversion to bf16
    f32_to_bf16<<<(CKK * CC + 255) / 256, 256, 0, stream>>>(Wq, Wqb, CKK * CC);
    f32_to_bf16<<<(CKK * CC + 255) / 256, 256, 0, stream>>>(Wk, Wkb, CKK * CC);
    f32_to_bf16<<<(CC * CC  + 255) / 256, 256, 0, stream>>>(Wv, Wvb, CC * CC);

    const dim3 blk(256);

    // 3) q = Wq * x + bq   -> stored transposed qT [B, N, Ck]
    wmma_gemm_bf16<<<dim3(NN / 64, CKK / 128, BB), blk, 0, stream>>>(
        Wqb, 0, CC,  xbT, (long long)NN * CC, CC,  bq,
        qT, (long long)NN * CKK, CKK, /*mode=*/2, /*K=*/CC);

    // 4) k = Wk * y + bk   -> stored transposed kT [B, N, Ck]
    wmma_gemm_bf16<<<dim3(NN / 64, CKK / 128, BB), blk, 0, stream>>>(
        Wkb, 0, CC,  ybT, (long long)NN * CC, CC,  bk,
        kT, (long long)NN * CKK, CKK, /*mode=*/2, /*K=*/CC);

    // 5) v = Wv * y + bv   -> bf16 row-major [B, C, N]
    wmma_gemm_bf16<<<dim3(NN / 64, CC / 128, BB), blk, 0, stream>>>(
        Wvb, 0, CC,  ybT, (long long)NN * CC, CC,  bv,
        vB, (long long)CC * NN, NN, /*mode=*/1, /*K=*/CC);

    // 6) energy[n,m] = sum_o qT[n,o] * kT[m,o]   (A=qT, Bt=kT) -> f32
    wmma_gemm_bf16<<<dim3(NN / 64, NN / 128, BB), blk, 0, stream>>>(
        qT, (long long)NN * CKK, CKK,  kT, (long long)NN * CKK, CKK,  nullptr,
        ener, (long long)NN * NN, NN, /*mode=*/0, /*K=*/CKK);

    // 7) attn = softmax(energy * Ck^-0.5) over m  -> bf16 [B*N, N]
    const float scale = 0.08838834764831845f;   // 128^-0.5
    softmax_row<<<BB * NN, 256, 0, stream>>>(ener, attn, scale);

    // 8) out[c,n] = sum_m v[c,m] * attn[n,m]   (A=v, Bt=attn) -> f32 [B,C,N]
    wmma_gemm_bf16<<<dim3(NN / 64, CC / 128, BB), blk, 0, stream>>>(
        vB, (long long)CC * NN, NN,  attn, (long long)NN * NN, NN,  nullptr,
        outp, (long long)CC * NN, NN, /*mode=*/0, /*K=*/NN);

    // 9) nearest x4 upsample + residual add with c2
    upsample_add<<<(BB * CC * HH * WW / 4) / 256, 256, 0, stream>>>(outp, c2, out);
}
